// SpikeMLP_74045236183459
// MI455X (gfx1250) — matrix operand order repaired and checked by reading, not compiled, not run on hardware
//
#include <hip/hip_runtime.h>

// CDNA5 / gfx1250 wave32 WMMA implementation of the BrainCog-style spiking MLP.
// Per timestep: bf16 WMMA GEMM -> fused popnorm+LIF.  Weights transpose-cast
// once per launch to bf16 [N][K] so A and B tiles both stage contiguous-in-K.
// GEMM uses double-buffered LDS with async global->LDS copies (ASYNCcnt) when
// the toolchain exposes them, 128x256 block tiles, 64x64 per-wave tiles.

#if __has_builtin(__builtin_amdgcn_global_load_async_to_lds_b128)
#define HAS_ASYNC_LDS 1
#else
#define HAS_ASYNC_LDS 0
#endif

typedef __attribute__((ext_vector_type(16))) __bf16 bf16x16;
typedef __attribute__((ext_vector_type(8)))  __bf16 bf16x8;
typedef __attribute__((ext_vector_type(4)))  __bf16 bf16x4;
typedef __attribute__((ext_vector_type(8)))  float  v8f;
typedef __attribute__((ext_vector_type(4)))  float  v4f;
typedef int v4i_g __attribute__((vector_size(16)));   // matches builtin proto

#define BM 128
#define BN 256
#define BK 32
#define LDSK 40   // padded LDS row length in bf16 elems (80 B, 16B-aligned)

__device__ __forceinline__ void async_b128(const void* g, void* l) {
#if HAS_ASYNC_LDS
  __builtin_amdgcn_global_load_async_to_lds_b128(
      (__attribute__((address_space(1))) v4i_g*)g,
      (__attribute__((address_space(3))) v4i_g*)l, 0, 0);
#endif
}

__device__ __forceinline__ void wait_async0() {
#if HAS_ASYNC_LDS
#if __has_builtin(__builtin_amdgcn_s_wait_asynccnt)
  __builtin_amdgcn_s_wait_asynccnt(0);
#else
  asm volatile("s_wait_asynccnt 0x0" ::: "memory");
#endif
#endif
}

// ---------------------------------------------------------------------------
// GEMM: C[M][N] (f32) = A[M][K] * Bt[N][K]^T, optional per-column bias.
// A_F32 != 0 : A is f32, converted (and scaled) to bf16 while staging.
// Requires M%BM==0, N%BN==0, K%BK==0 (true for all calls here).
// ---------------------------------------------------------------------------
template<int A_F32>
__global__ __launch_bounds__(256)
void gemm_wmma_bf16(const void* __restrict__ Asrc, const __bf16* __restrict__ Bt,
                    float* __restrict__ C, const float* __restrict__ bias,
                    int M, int N, int K, float ascale)
{
  __shared__ __align__(16) __bf16 sA[2][BM * LDSK];   // 2 x 10240 B
  __shared__ __align__(16) __bf16 sB[2][BN * LDSK];   // 2 x 20480 B

  const int tid  = threadIdx.x;
  const int wave = tid >> 5, lane = tid & 31;
  const int lr   = lane & 15, hi = lane >> 4;
  const int wm   = wave & 1;    // 2 wave-rows  (64 M each)
  const int wn   = wave >> 1;   // 4 wave-cols  (64 N each)
  const int m0 = blockIdx.y * BM;
  const int n0 = blockIdx.x * BN;

  const float*  Af = (const float*)Asrc;
  const __bf16* Ab = (const __bf16*)Asrc;

  // manual-staging registers (unused paths optimize away)
  v4f    areg[4];   // A f32 path: 4 chunks of 4 floats
  bf16x8 abreg[2];  // A bf16 fallback: 2 chunks of 16 B
  bf16x8 breg[4];   // B fallback: 4 chunks of 16 B

  auto stage_issue = [&](int k0, int buf) {
    if (A_F32) {
#pragma unroll
      for (int i = 0; i < 4; ++i) {
        const int c = tid + (i << 8);                 // 128 rows x 8 chunks
        areg[i] = *(const v4f*)(Af + (size_t)(m0 + (c >> 3)) * K + k0 + ((c & 7) << 2));
      }
    } else if (HAS_ASYNC_LDS) {
#pragma unroll
      for (int i = 0; i < 2; ++i) {
        const int c = tid + (i << 8);                 // 128 rows x 4 chunks
        async_b128(Ab + (size_t)(m0 + (c >> 2)) * K + k0 + ((c & 3) << 3),
                   &sA[buf][(c >> 2) * LDSK + ((c & 3) << 3)]);
      }
    } else {
#pragma unroll
      for (int i = 0; i < 2; ++i) {
        const int c = tid + (i << 8);
        abreg[i] = *(const bf16x8*)(Ab + (size_t)(m0 + (c >> 2)) * K + k0 + ((c & 3) << 3));
      }
    }
    if (HAS_ASYNC_LDS) {
#pragma unroll
      for (int i = 0; i < 4; ++i) {
        const int c = tid + (i << 8);                 // 256 rows x 4 chunks
        async_b128(Bt + (size_t)(n0 + (c >> 2)) * K + k0 + ((c & 3) << 3),
                   &sB[buf][(c >> 2) * LDSK + ((c & 3) << 3)]);
      }
    } else {
#pragma unroll
      for (int i = 0; i < 4; ++i) {
        const int c = tid + (i << 8);
        breg[i] = *(const bf16x8*)(Bt + (size_t)(n0 + (c >> 2)) * K + k0 + ((c & 3) << 3));
      }
    }
  };

  auto stage_commit = [&](int buf) {
    if (A_F32) {
#pragma unroll
      for (int i = 0; i < 4; ++i) {
        const int c = tid + (i << 8);
        bf16x4 vb;
#pragma unroll
        for (int e = 0; e < 4; ++e) vb[e] = (__bf16)(areg[i][e] * ascale);
        *(bf16x4*)(&sA[buf][(c >> 3) * LDSK + ((c & 7) << 2)]) = vb;
      }
    } else if (!HAS_ASYNC_LDS) {
#pragma unroll
      for (int i = 0; i < 2; ++i) {
        const int c = tid + (i << 8);
        *(bf16x8*)(&sA[buf][(c >> 2) * LDSK + ((c & 3) << 3)]) = abreg[i];
      }
    }
    if (!HAS_ASYNC_LDS) {
#pragma unroll
      for (int i = 0; i < 4; ++i) {
        const int c = tid + (i << 8);
        *(bf16x8*)(&sB[buf][(c >> 2) * LDSK + ((c & 3) << 3)]) = breg[i];
      }
    }
  };

  v8f acc[4][4];
#pragma unroll
  for (int i = 0; i < 4; ++i)
#pragma unroll
    for (int j = 0; j < 4; ++j) acc[i][j] = (v8f){};

  stage_issue(0, 0);
  stage_commit(0);
  wait_async0();
  __syncthreads();

  int buf = 0;
  for (int k0 = 0; k0 < K; k0 += BK) {
    const int  nxt      = buf ^ 1;
    const bool havenext = (k0 + BK) < K;
    if (havenext) stage_issue(k0 + BK, nxt);   // overlaps with WMMAs below

    // B fragments (16-bit 32x16): lanes 0-15 hold K 0..15, lanes 16-31 K 16..31
    bf16x16 bfr[4];
    const int khb = hi << 4;
#pragma unroll
    for (int ni = 0; ni < 4; ++ni) {
      const int r = wn * 64 + ni * 16 + lr;
      union { bf16x16 v; bf16x8 h[2]; } u;
      u.h[0] = *(const bf16x8*)(&sB[buf][r * LDSK + khb]);
      u.h[1] = *(const bf16x8*)(&sB[buf][r * LDSK + khb + 8]);
      bfr[ni] = u.v;
    }
    // A fragments (16-bit 16x32): lanes 0-15 hold K 0..15, lanes 16-31 K 16..31
    const int kha = hi << 4;
#pragma unroll
    for (int mi = 0; mi < 4; ++mi) {
      const int r = wm * 64 + mi * 16 + lr;
      union { bf16x16 v; bf16x8 h[2]; } u;
      u.h[0] = *(const bf16x8*)(&sA[buf][r * LDSK + kha]);
      u.h[1] = *(const bf16x8*)(&sA[buf][r * LDSK + kha + 8]);
      const bf16x16 afr = u.v;
#pragma unroll
      for (int ni = 0; ni < 4; ++ni)
        acc[mi][ni] = __builtin_amdgcn_wmma_f32_16x16x32_bf16(
            false, afr, false, bfr[ni], (short)0, acc[mi][ni], false, false);
    }

    if (havenext) {
      stage_commit(nxt);
      wait_async0();
      __syncthreads();
    }
    buf = nxt;
  }

  // Epilogue: D layout = lanes 0-15 rows v, lanes 16-31 rows v+8, col = lane&15
#pragma unroll
  for (int mi = 0; mi < 4; ++mi) {
#pragma unroll
    for (int ni = 0; ni < 4; ++ni) {
      const int rbase = m0 + wm * 64 + mi * 16 + hi * 8;
      const int col   = n0 + wn * 64 + ni * 16 + lr;
      const float bb  = bias ? bias[col] : 0.0f;
#pragma unroll
      for (int v = 0; v < 8; ++v)
        C[(size_t)(rbase + v) * N + col] = acc[mi][ni][v] + bb;
    }
  }
}

// ---------------------------------------------------------------------------
// Fused popnorm (threshold-scaled layernorm over U) + LIF update.
// One 256-thread block per batch row; U%256==0, U<=2048.
// ---------------------------------------------------------------------------
__global__ __launch_bounds__(256)
void popnorm_lif_kernel(const float* __restrict__ H,
                        const float* __restrict__ g, const float* __restrict__ b,
                        float* __restrict__ mem, __bf16* __restrict__ spk,
                        float* __restrict__ acc, int U, int do_acc)
{
  __shared__ float r1[256], r2[256];
  const int tid = threadIdx.x;
  const size_t base = (size_t)blockIdx.x * U;
  const int nit = U >> 8;

  float hv[8];
  float s = 0.0f, ss = 0.0f;
  for (int it = 0; it < nit; ++it) {
    const float h = H[base + tid + it * 256];
    hv[it] = h; s += h; ss += h * h;
  }
  r1[tid] = s; r2[tid] = ss;
  __syncthreads();
  for (int off = 128; off > 0; off >>= 1) {
    if (tid < off) { r1[tid] += r1[tid + off]; r2[tid] += r2[tid + off]; }
    __syncthreads();
  }
  const float invU = 1.0f / (float)U;
  const float mean = r1[0] * invU;
  const float var  = r2[0] * invU - mean * mean;
  const float rstd = rsqrtf(var + 1e-5f);

  for (int it = 0; it < nit; ++it) {
    const int j = tid + it * 256;
    const float y = 0.5f * ((hv[it] - mean) * rstd * g[j] + b[j]);  // THR=0.5
    float m = mem[base + j];
    m = 0.5f * (m + y);                      // mem += (y - mem)/TAU, TAU=2
    const float sp = (m > 0.5f) ? 1.0f : 0.0f;
    m = (sp > 0.0f) ? 0.0f : m;              // hard reset
    mem[base + j] = m;
    spk[base + j] = (__bf16)sp;
    if (do_acc) acc[base + j] += sp;
  }
}

// ---------------------------------------------------------------------------
// Transpose + cast: in f32 [K][N] -> out bf16 [N][K].  K%32==0, N%32==0.
// ---------------------------------------------------------------------------
__global__ __launch_bounds__(256)
void transpose_cast_kernel(const float* __restrict__ in, __bf16* __restrict__ out,
                           int K, int N)
{
  __shared__ float tile[32][33];
  const int nb = blockIdx.x * 32, kb = blockIdx.y * 32;
  const int tx = threadIdx.x & 31, ty = threadIdx.x >> 5;   // 32x8
#pragma unroll
  for (int i = 0; i < 32; i += 8)
    tile[ty + i][tx] = in[(size_t)(kb + ty + i) * N + nb + tx];
  __syncthreads();
#pragma unroll
  for (int i = 0; i < 32; i += 8)
    out[(size_t)(nb + ty + i) * K + kb + tx] = (__bf16)tile[tx][ty + i];
}

// ---------------------------------------------------------------------------
extern "C" void kernel_launch(void* const* d_in, const int* in_sizes, int n_in,
                              void* d_out, int out_size, void* d_ws, size_t ws_size,
                              hipStream_t stream)
{
  const int T = 5, B = 4096, D = 2048, U = 2048, O = 1024;
  const float* feat = (const float*)d_in[0];
  const float* W0 = (const float*)d_in[1];
  const float* g0 = (const float*)d_in[2];
  const float* b0 = (const float*)d_in[3];
  const float* W1 = (const float*)d_in[4];
  const float* g1 = (const float*)d_in[5];
  const float* b1 = (const float*)d_in[6];
  const float* W2 = (const float*)d_in[7];
  const float* g2 = (const float*)d_in[8];
  const float* b2 = (const float*)d_in[9];
  const float* Wm = (const float*)d_in[10];
  const float* bm = (const float*)d_in[11];
  float* out = (float*)d_out;
  (void)in_sizes; (void)n_in; (void)out_size; (void)ws_size;

  // Workspace carve-out (~204 MB total)
  char* ws = (char*)d_ws;
  size_t off = 0;
  auto carve = [&](size_t bytes) -> void* {
    void* p = ws + off;
    off += (bytes + 255) & ~(size_t)255;
    return p;
  };
  __bf16* W0t = (__bf16*)carve((size_t)D * U * 2);
  __bf16* W1t = (__bf16*)carve((size_t)U * U * 2);
  __bf16* W2t = (__bf16*)carve((size_t)U * U * 2);
  __bf16* Wmt = (__bf16*)carve((size_t)U * O * 2);
  float*  H   = (float*) carve((size_t)B * U * 4);
  float*  m0  = (float*) carve((size_t)B * U * 4);
  float*  m1  = (float*) carve((size_t)B * U * 4);
  float*  m2  = (float*) carve((size_t)B * U * 4);
  float*  acc = (float*) carve((size_t)B * U * 4);
  __bf16* spk = (__bf16*)carve((size_t)B * U * 2);

  // Zero persistent state every launch (deterministic; graph-capture safe)
  (void)hipMemsetAsync(m0,  0, (size_t)B * U * 4, stream);
  (void)hipMemsetAsync(m1,  0, (size_t)B * U * 4, stream);
  (void)hipMemsetAsync(m2,  0, (size_t)B * U * 4, stream);
  (void)hipMemsetAsync(acc, 0, (size_t)B * U * 4, stream);

  // Transpose-cast weights to bf16 [N][K]
  transpose_cast_kernel<<<dim3(U / 32, D / 32), 256, 0, stream>>>(W0, W0t, D, U);
  transpose_cast_kernel<<<dim3(U / 32, U / 32), 256, 0, stream>>>(W1, W1t, U, U);
  transpose_cast_kernel<<<dim3(U / 32, U / 32), 256, 0, stream>>>(W2, W2t, U, U);
  transpose_cast_kernel<<<dim3(O / 32, U / 32), 256, 0, stream>>>(Wm, Wmt, U, O);

  const dim3 blk(256);
  const dim3 gUU(U / BN, B / BM);   // (8, 32) blocks
  const dim3 gUO(O / BN, B / BM);   // (4, 32) blocks

  for (int t = 0; t < T; ++t) {
    gemm_wmma_bf16<1><<<gUU, blk, 0, stream>>>(feat + (size_t)t * B * D, W0t, H,
                                               nullptr, B, U, D, 1.0f);
    popnorm_lif_kernel<<<B, 256, 0, stream>>>(H, g0, b0, m0, spk, acc, U, 0);
    gemm_wmma_bf16<0><<<gUU, blk, 0, stream>>>(spk, W1t, H, nullptr, B, U, U, 1.0f);
    popnorm_lif_kernel<<<B, 256, 0, stream>>>(H, g1, b1, m1, spk, acc, U, 0);
    gemm_wmma_bf16<0><<<gUU, blk, 0, stream>>>(spk, W2t, H, nullptr, B, U, U, 1.0f);
    popnorm_lif_kernel<<<B, 256, 0, stream>>>(H, g2, b2, m2, spk, acc, U, 1);
  }
  // Final projection: (acc/T) @ Wm + bm  (1/T applied during bf16 staging)
  gemm_wmma_bf16<1><<<gUO, blk, 0, stream>>>(acc, Wmt, out, bm, B, O, U, 0.2f);
}
